// MultiHeadAttention_36051955483000
// MI455X (gfx1250) — matrix operand order repaired and checked by reading, not compiled, not run on hardware
//
#include <hip/hip_runtime.h>
#include <hip/hip_bf16.h>
#include <stdint.h>

// ---------------------------------------------------------------------------
// Causal MHA forward for MI455X (gfx1250, wave32, WMMA).
// Pipeline: [weights -> bf16 transpose] -> [QKV proj GEMM] -> [flash attention
// with double-buffered GLOBAL_LOAD_ASYNC_TO_LDS K/V staging] -> [output GEMM].
// All matrix math on v_wmma_f32_16x16x32_bf16.
// ---------------------------------------------------------------------------

#define DEV static __device__ __forceinline__

typedef __attribute__((ext_vector_type(16))) __bf16          v16bf;
typedef __attribute__((ext_vector_type(8)))  float           v8f;
typedef __attribute__((ext_vector_type(16))) unsigned short  v16u;
typedef __attribute__((ext_vector_type(8)))  unsigned short  v8u;

constexpr int   Bc = 4, Nc = 2048, Dc = 1024, Hc = 16, DHc = 64;
constexpr float SCALE     = 0.125f;     // dh^-0.5, dh = 64
constexpr float MASK_FILL = -32767.f;   // matches reference

// float -> bf16 round-to-nearest-even
DEV unsigned short f2bf(float f) {
  unsigned u = __builtin_bit_cast(unsigned, f);
  u += 0x7FFFu + ((u >> 16) & 1u);
  return (unsigned short)(u >> 16);
}

DEV v16bf as_bf(v16u u) { return __builtin_bit_cast(v16bf, u); }

// B fragment (K=32 x N=16 bf16): lane L holds column N=L&15; elems 0..7 are
// K = [8ha, +8), elems 8..15 are K = [16+8ha, +8) (ld = K stride).
DEV v16bf loadB(const unsigned short* col, int ld, int ha) {
  v16u u;
#pragma unroll
  for (int e = 0; e < 8; ++e) {
    u[e] = col[(size_t)(8 * ha + e) * ld];
    u[e + 8] = col[(size_t)(16 + 8 * ha + e) * ld];
  }
  return as_bf(u);
}

// A fragment (M=16 x K=32 bf16) per ISA layout: lane L -> row L&15, ha=L>>4;
// elems 0..7 are K = [k0+8ha, +8), elems 8..15 are K = [k0+16+8ha, +8).
union AFrag { v16u v; v8u h[2]; };
DEV v16bf loadA_bf(const unsigned short* row, int k0, int ha) {
  AFrag a;
  a.h[0] = *(const v8u*)(row + k0 + 8 * ha);
  a.h[1] = *(const v8u*)(row + k0 + 16 + 8 * ha);
  return as_bf(a.v);
}

DEV v16bf loadA_f32(const float* row, int k0, int ha) {
  v8f f0 = *(const v8f*)(row + k0 + 8 * ha);
  v8f f1 = *(const v8f*)(row + k0 + 16 + 8 * ha);
  v16u u;
#pragma unroll
  for (int e = 0; e < 8; ++e) { u[e] = f2bf(f0[e]); u[e + 8] = f2bf(f1[e]); }
  return as_bf(u);
}

DEV v8f wmma_bf16(v16bf a, v16bf b, v8f c) {
  return __builtin_amdgcn_wmma_f32_16x16x32_bf16(false, a, false, b, (short)0, c,
                                                 false, false);
}

// ---- CDNA5 async copy to LDS (ASYNCcnt path, CDNA5 ISA) ----
// Generic pointer to a __shared__ object is {shared_aperture_hi, lds_offset};
// truncation yields the wave-relative LDS byte offset the instruction wants
// (the VDST VGPR supplies this LDS byte offset).
DEV unsigned ldsoff(const void* p) { return (unsigned)(size_t)p; }

DEV void async_b128(const void* g, unsigned l) {
  asm volatile("global_load_async_to_lds_b128 %0, %1, off"
               :: "v"(l), "v"((unsigned long long)(size_t)g)
               : "memory");
}

DEV void wait_asynccnt0() { asm volatile("s_wait_asynccnt 0x0" ::: "memory"); }

// ---------------------------------------------------------------------------
// Kernel 1: W (out,in) f32 -> WT (in,out) bf16, 32x32 LDS tile transpose.
// ---------------------------------------------------------------------------
__global__ __launch_bounds__(256) void wt_kernel(const float* __restrict__ W,
                                                 unsigned short* __restrict__ WT) {
  __shared__ float tile[32][33];
  const int k0 = blockIdx.x * 32, n0 = blockIdx.y * 32;
  const int tx = threadIdx.x, ty = threadIdx.y;  // 32 x 8
#pragma unroll
  for (int i = 0; i < 4; ++i)
    tile[ty + 8 * i][tx] = W[(size_t)(n0 + ty + 8 * i) * Dc + k0 + tx];
  __syncthreads();
#pragma unroll
  for (int i = 0; i < 4; ++i)
    WT[(size_t)(k0 + ty + 8 * i) * Dc + n0 + tx] = f2bf(tile[tx][ty + 8 * i]);
}

// ---------------------------------------------------------------------------
// Kernel 2: QKV projection.  One wave -> 16 rows x 64 cols output strip.
// blockIdx.x = head (64-col tile), blockIdx.y = 128-row block, blockIdx.z = q/k/v.
// Q scaled by dh^-0.5 -> [b,h,n,dh]; K transposed -> [b,h,dh,n]; V -> [b,h,n,dh].
// ---------------------------------------------------------------------------
__global__ __launch_bounds__(256) void qkv_proj_kernel(
    const float* __restrict__ q, const float* __restrict__ k,
    const float* __restrict__ v, const unsigned short* __restrict__ WTq,
    const unsigned short* __restrict__ WTk, const unsigned short* __restrict__ WTv,
    const float* __restrict__ bq, const float* __restrict__ bk,
    const float* __restrict__ bv, unsigned short* __restrict__ Qh,
    unsigned short* __restrict__ KhT, unsigned short* __restrict__ Vh) {
  const int sel = blockIdx.z;
  const float* X = sel == 0 ? q : sel == 1 ? k : v;
  const unsigned short* WT = sel == 0 ? WTq : sel == 1 ? WTk : WTv;
  const float* bias = sel == 0 ? bq : sel == 1 ? bk : bv;

  const int lane = threadIdx.x & 31, w = threadIdx.x >> 5;
  const int ha = lane >> 4, ln = lane & 15;
  const int r0 = blockIdx.y * 128 + w * 16;  // global row (b*N + n)
  const int o0 = blockIdx.x * 64;            // output col = head*64

  v8f acc[4] = {};
  const float* xrow = X + (size_t)(r0 + ln) * Dc;
  for (int k0 = 0; k0 < Dc; k0 += 32) {
    v16bf a = loadA_f32(xrow, k0, ha);
#pragma unroll
    for (int t = 0; t < 4; ++t) {
      v16bf bf = loadB(WT + (size_t)k0 * Dc + o0 + 16 * t + ln, Dc, ha);
      acc[t] = wmma_bf16(a, bf, acc[t]);
    }
  }
  const int b = r0 / Nc, n = r0 % Nc, h = blockIdx.x;
#pragma unroll
  for (int t = 0; t < 4; ++t) {
    const float bval = bias[o0 + 16 * t + ln];
#pragma unroll
    for (int j = 0; j < 8; ++j) {
      const int nn = n + j + 8 * ha;  // C/D layout: row = j + 8*(lane>=16)
      const int dd = 16 * t + ln;     // col = lane & 15
      const float val = acc[t][j] + bval;
      if (sel == 0)
        Qh[(((size_t)(b * Hc + h)) * Nc + nn) * DHc + dd] = f2bf(val * SCALE);
      else if (sel == 1)
        KhT[(((size_t)(b * Hc + h)) * DHc + dd) * Nc + nn] = f2bf(val);
      else
        Vh[(((size_t)(b * Hc + h)) * Nc + nn) * DHc + dd] = f2bf(val);
    }
  }
}

// ---------------------------------------------------------------------------
// Kernel 3: flash-style causal attention.
// Block-uniform key loop; each 32-key K tile (64x32) and V tile (32x64) is
// staged into LDS once per block via double-buffered async copies, overlapping
// the next tile's DMA with the current tile's WMMAs.  One wave = 16 query rows;
// tiles past a wave's causal bound are fully masked (contribute exactly 0).
// ---------------------------------------------------------------------------
__global__ __launch_bounds__(256) void attn_kernel(
    const unsigned short* __restrict__ Qh, const unsigned short* __restrict__ KhT,
    const unsigned short* __restrict__ Vh, unsigned short* __restrict__ Xmid) {
  __shared__ unsigned short Kt[2][64][32];  // [buf][d][m_local]
  __shared__ unsigned short Vt[2][32][64];  // [buf][m_local][d]
  __shared__ unsigned short pt[8][16][40];  // per-wave P-transpose tile

  const int tid = threadIdx.x;
  const int lane = tid & 31, w = tid >> 5;
  const int ha = lane >> 4, ln = lane & 15;
  const int bh = blockIdx.x;  // b*H + h
  const int by = blockIdx.y;
  const int n0 = by * 128 + w * 16;  // this wave's query rows

  const unsigned short* Qb = Qh + (size_t)bh * Nc * DHc;
  const unsigned short* Kb = KhT + (size_t)bh * DHc * Nc;
  const unsigned short* Vb = Vh + (size_t)bh * Nc * DHc;

  // Preload Q A-fragments (d = 0..31, 32..63); softmax scale already folded in.
  const unsigned short* qrow = Qb + (size_t)(n0 + ln) * DHc;
  const v16bf aq0 = loadA_bf(qrow, 0, ha);
  const v16bf aq1 = loadA_bf(qrow, 32, ha);

  // Cooperative staging split across the 256 threads (16B chunk per thread per
  // tile): K tile 64 rows x 64B, V tile 32 rows x 128B.
  const int krow = tid >> 2, kcol = (tid & 3) * 8;
  const int vrow = tid >> 3, vcol = (tid & 7) * 8;
  auto stage = [&](int mt, int buf) {
    async_b128(Kb + (size_t)krow * Nc + mt * 32 + kcol,
               ldsoff(&Kt[buf][krow][kcol]));
    async_b128(Vb + (size_t)(mt * 32 + vrow) * DHc + vcol,
               ldsoff(&Vt[buf][vrow][vcol]));
  };

  v8f o[4] = {};  // o[t][j] = O[n0+j+8ha][16t+ln]
  float runm[8], runl[8];
#pragma unroll
  for (int j = 0; j < 8; ++j) { runm[j] = -1e30f; runl[j] = 0.f; }

  const int mtiles = 4 * by + 4;  // covers keys 0 .. by*128+127
  stage(0, 0);
  for (int mt = 0; mt < mtiles; ++mt) {
    const int buf = mt & 1;
    const int m0 = mt * 32;
    wait_asynccnt0();   // own wave's staged chunks have landed in LDS
    __syncthreads();    // everyone's chunks visible block-wide
    if (mt + 1 < mtiles) stage(mt + 1, buf ^ 1);

    // ---- S = Q K^T (16x32 tile as two 16x16 D frags), accumulate over dh=64
    v8f s0 = {}, s1 = {};
    {
      v16bf b00 = loadB(&Kt[buf][0][ln], 32, ha);       // d=0..31,  m=0..15
      v16bf b01 = loadB(&Kt[buf][32][ln], 32, ha);      // d=32..63, m=0..15
      v16bf b10 = loadB(&Kt[buf][0][16 + ln], 32, ha);  // d=0..31,  m=16..31
      v16bf b11 = loadB(&Kt[buf][32][16 + ln], 32, ha); // d=32..63, m=16..31
      s0 = wmma_bf16(aq0, b00, s0);
      s0 = wmma_bf16(aq1, b01, s0);
      s1 = wmma_bf16(aq0, b10, s1);
      s1 = wmma_bf16(aq1, b11, s1);
    }
    // ---- causal mask (boundary/past-bound tiles; interior tiles unmasked)
    if (m0 + 31 > n0) {
#pragma unroll
      for (int j = 0; j < 8; ++j) {
        const int nn = n0 + j + 8 * ha;
        if (m0 + ln > nn)      s0[j] = MASK_FILL;
        if (m0 + 16 + ln > nn) s1[j] = MASK_FILL;
      }
    }
    // ---- online softmax: rows live in-lane (replicated over 16-lane halves)
#pragma unroll
    for (int j = 0; j < 8; ++j) {
      float m = fmaxf(s0[j], s1[j]);
#pragma unroll
      for (int d = 1; d < 16; d <<= 1) m = fmaxf(m, __shfl_xor(m, d));
      const float newm = fmaxf(runm[j], m);
      const float al = __expf(runm[j] - newm);
      const float p0 = __expf(s0[j] - newm);
      const float p1 = __expf(s1[j] - newm);
      float rs = p0 + p1;
#pragma unroll
      for (int d = 1; d < 16; d <<= 1) rs += __shfl_xor(rs, d);
      runl[j] = runl[j] * al + rs;
      runm[j] = newm;
#pragma unroll
      for (int t = 0; t < 4; ++t) o[t][j] *= al;
      // D-frag (row j+8ha, col ln / 16+ln) -> LDS for A-frag transpose
      pt[w][j + 8 * ha][ln] = f2bf(p0);
      pt[w][j + 8 * ha][16 + ln] = f2bf(p1);
    }
    __builtin_amdgcn_wave_barrier();  // ds_store before ds_load (per-wave tile)
    const v16bf pa = loadA_bf(&pt[w][ln][0], 0, ha);
    __builtin_amdgcn_wave_barrier();
    // ---- O += P @ V  (K = 32 keys, four 16-col d tiles)
#pragma unroll
    for (int t = 0; t < 4; ++t) {
      v16bf bv = loadB(&Vt[buf][0][16 * t + ln], 64, ha);
      o[t] = wmma_bf16(pa, bv, o[t]);
    }
    __syncthreads();  // all compute on `buf` done before it is re-staged
  }
  // ---- finalize: divide by row sums, write bf16 [b, n, h*64+d]
  const int b = bh / Hc, h = bh % Hc;
#pragma unroll
  for (int t = 0; t < 4; ++t)
#pragma unroll
    for (int j = 0; j < 8; ++j) {
      const int nn = n0 + j + 8 * ha;
      Xmid[((size_t)(b * Nc + nn)) * Dc + h * DHc + 16 * t + ln] =
          f2bf(o[t][j] / runl[j]);
    }
}

// ---------------------------------------------------------------------------
// Kernel 4: output projection (bf16 GEMM, no bias), f32 result to d_out.
// ---------------------------------------------------------------------------
__global__ __launch_bounds__(256) void out_proj_kernel(
    const unsigned short* __restrict__ Xmid, const unsigned short* __restrict__ WTo,
    float* __restrict__ out) {
  const int lane = threadIdx.x & 31, w = threadIdx.x >> 5;
  const int ha = lane >> 4, ln = lane & 15;
  const int r0 = blockIdx.y * 128 + w * 16;
  const int o0 = blockIdx.x * 64;
  v8f acc[4] = {};
  const unsigned short* xrow = Xmid + (size_t)(r0 + ln) * Dc;
  for (int k0 = 0; k0 < Dc; k0 += 32) {
    v16bf a = loadA_bf(xrow, k0, ha);
#pragma unroll
    for (int t = 0; t < 4; ++t) {
      v16bf bf = loadB(WTo + (size_t)k0 * Dc + o0 + 16 * t + ln, Dc, ha);
      acc[t] = wmma_bf16(a, bf, acc[t]);
    }
  }
#pragma unroll
  for (int t = 0; t < 4; ++t)
#pragma unroll
    for (int j = 0; j < 8; ++j)
      out[(size_t)(r0 + j + 8 * ha) * Dc + o0 + 16 * t + ln] = acc[t][j];
}

// ---------------------------------------------------------------------------
extern "C" void kernel_launch(void* const* d_in, const int* in_sizes, int n_in,
                              void* d_out, int out_size, void* d_ws, size_t ws_size,
                              hipStream_t stream) {
  const float* q  = (const float*)d_in[0];
  const float* k  = (const float*)d_in[1];
  const float* v  = (const float*)d_in[2];
  const float* Wq = (const float*)d_in[3];
  const float* bq = (const float*)d_in[4];
  const float* Wk = (const float*)d_in[5];
  const float* bk = (const float*)d_in[6];
  const float* Wv = (const float*)d_in[7];
  const float* bv = (const float*)d_in[8];
  const float* Wo = (const float*)d_in[9];
  float* out = (float*)d_out;

  uint8_t* ws = (uint8_t*)d_ws;
  size_t off = 0;
  auto alloc = [&](size_t bytes) {
    void* p = ws + off;
    off += (bytes + 255) & ~(size_t)255;
    return p;
  };
  const size_t WB = (size_t)Dc * Dc * 2;             // 2 MB each
  const size_t HB = (size_t)Bc * Hc * Nc * DHc * 2;  // 16 MB each
  unsigned short* WTq  = (unsigned short*)alloc(WB);
  unsigned short* WTk  = (unsigned short*)alloc(WB);
  unsigned short* WTv  = (unsigned short*)alloc(WB);
  unsigned short* WTo  = (unsigned short*)alloc(WB);
  unsigned short* Qh   = (unsigned short*)alloc(HB);
  unsigned short* KhT  = (unsigned short*)alloc(HB);
  unsigned short* Vh   = (unsigned short*)alloc(HB);
  unsigned short* Xmid = (unsigned short*)alloc((size_t)Bc * Nc * Dc * 2);

  const dim3 tb(32, 8);
  const dim3 tg(Dc / 32, Dc / 32);
  wt_kernel<<<tg, tb, 0, stream>>>(Wq, WTq);
  wt_kernel<<<tg, tb, 0, stream>>>(Wk, WTk);
  wt_kernel<<<tg, tb, 0, stream>>>(Wv, WTv);
  wt_kernel<<<tg, tb, 0, stream>>>(Wo, WTo);

  qkv_proj_kernel<<<dim3(Dc / 64, (Bc * Nc) / 128, 3), 256, 0, stream>>>(
      q, k, v, WTq, WTk, WTv, bq, bk, bv, Qh, KhT, Vh);

  attn_kernel<<<dim3(Bc * Hc, Nc / 128), 256, 0, stream>>>(Qh, KhT, Vh, Xmid);

  out_proj_kernel<<<dim3(Dc / 64, (Bc * Nc) / 128), 256, 0, stream>>>(Xmid, WTo, out);
}
